// MultiScaleRetention_31671088841057
// MI455X (gfx1250) — compile-verified, matrix operand order checked by reading, not run on hardware
//
#include <hip/hip_runtime.h>
#include <hip/hip_bf16.h>
#include <stdint.h>

#define B 2
#define T 2048
#define HID 2048
#define H 8
#define DK 256
#define DV 512
#define NCHUNK 32

typedef __bf16 bf16;
typedef __attribute__((ext_vector_type(16))) __bf16 v16bf;
typedef __attribute__((ext_vector_type(8)))  __bf16 bf16x8;
typedef __attribute__((ext_vector_type(8)))  float  v8f;

__device__ __forceinline__ bf16 f2bf(float f) {
  union { float f; uint32_t u; } v; v.f = f;
  uint32_t r = v.u + 0x7FFFu + ((v.u >> 16) & 1u);
  uint16_t h = (uint16_t)(r >> 16);
  return __builtin_bit_cast(bf16, h);
}

// Async DMA: global -> LDS, 16 bytes per lane, tracked by ASYNCcnt.
__device__ __forceinline__ void async_b128(uint32_t lds_off, const void* gptr) {
  uint64_t ga = (uint64_t)(uintptr_t)gptr;
  asm volatile("global_load_async_to_lds_b128 %0, %1, off"
               :: "v"(lds_off), "v"(ga) : "memory");
}
__device__ __forceinline__ void wait_async0() {
  asm volatile("s_wait_asynccnt 0" ::: "memory");
}
__device__ __forceinline__ uint32_t lds_addr(const void* p) {
  return (uint32_t)(uintptr_t)p;   // low 32 bits of generic shared ptr = LDS offset
}

// Build a 16x16x32 bf16 A/B fragment from a contiguous per-row K run.
// Pair p holds K = base(p)+{0,1}; base = (p<4?0:16) + 2*(p%4) + 8*half
// => elements [0..7] = K[8*half .. +7], [8..15] = K[16+8*half .. +7].
__device__ __forceinline__ v16bf frag_ld(const bf16* row, int kbase, int hlf) {
  bf16x8 lo = *(const bf16x8*)(row + kbase + 8 * hlf);
  bf16x8 hi = *(const bf16x8*)(row + kbase + 16 + 8 * hlf);
  v16bf r;
#pragma unroll
  for (int e = 0; e < 8; ++e) { r[e] = lo[e]; r[e + 8] = hi[e]; }
  return r;
}

__device__ __forceinline__ v8f wmma_bf(v16bf a, v16bf b, v8f c) {
  return __builtin_amdgcn_wmma_f32_16x16x32_bf16(false, a, false, b, (short)0, c,
                                                 false, false);
}

// ---------------- prep: fp32 -> bf16 ----------------
__global__ __launch_bounds__(256) void cvt_f2bf_kernel(const float* __restrict__ in,
                                                       bf16* __restrict__ out, int n) {
  int i = blockIdx.x * 256 + threadIdx.x;
  if (i < n) out[i] = f2bf(in[i]);
}

// Wt[n][k] = W[k][n]   (W is [K][N] row-major)
__global__ __launch_bounds__(256) void transpose_cvt_kernel(const float* __restrict__ W,
                                                            bf16* __restrict__ Wt,
                                                            int K, int N) {
  int idx = blockIdx.x * 256 + threadIdx.x;
  if (idx < K * N) {
    int n = idx / K;
    int k = idx - n * K;
    Wt[idx] = f2bf(W[(size_t)k * N + n]);
  }
}

// ---------------- generic bf16 WMMA GEMM: C[M][N] = A[M][K] * Bt[N][K]^T ----------------
// 128x128 block tile, async-DMA double-buffered K tiles of 32.
__global__ __launch_bounds__(256) void gemm_bf16_kernel(const bf16* __restrict__ A,
                                                        const bf16* __restrict__ Bt,
                                                        float* __restrict__ C,
                                                        int M, int N, int K) {
  __shared__ bf16 As[2][128][40];  // 32 K + 8 pad
  __shared__ bf16 Bs[2][128][40];
  const int tid = threadIdx.x;
  const int wave = tid >> 5, lane = tid & 31;
  const int hlf = lane >> 4, l16 = lane & 15;
  const int bm = blockIdx.y * 128, bn = blockIdx.x * 128;
  const int wm = (wave >> 2) * 64;   // wave row (0/64)
  const int wn = (wave & 3) * 32;    // wave col (0..96)

  v8f acc[4][2];
#pragma unroll
  for (int fm = 0; fm < 4; ++fm)
#pragma unroll
    for (int fn = 0; fn < 2; ++fn) acc[fm][fn] = (v8f){0,0,0,0,0,0,0,0};

  // staging map: 512 x 16B chunks per matrix; thread does chunks tid, tid+256.
  const bf16* gA = A + (size_t)bm * K;
  const bf16* gB = Bt + (size_t)bn * K;

  auto stage = [&](int buf, int k0) {
#pragma unroll
    for (int s = 0; s < 2; ++s) {
      int ch = tid + s * 256;
      int r = ch >> 2;             // 0..127
      int c8 = (ch & 3) * 8;       // 0,8,16,24 (bf16 elems)
      async_b128(lds_addr(&As[buf][r][c8]), gA + (size_t)r * K + k0 + c8);
      async_b128(lds_addr(&Bs[buf][r][c8]), gB + (size_t)r * K + k0 + c8);
    }
  };

  stage(0, 0);
  wait_async0();
  __syncthreads();

  int buf = 0;
  for (int k0 = 0; k0 < K; k0 += 32) {
    if (k0 + 32 < K) stage(buf ^ 1, k0 + 32);

    v16bf bfrag[2];
#pragma unroll
    for (int fn = 0; fn < 2; ++fn)
      bfrag[fn] = frag_ld(&Bs[buf][wn + fn * 16 + l16][0], 0, hlf);
#pragma unroll
    for (int fm = 0; fm < 4; ++fm) {
      v16bf afrag = frag_ld(&As[buf][wm + fm * 16 + l16][0], 0, hlf);
#pragma unroll
      for (int fn = 0; fn < 2; ++fn)
        acc[fm][fn] = wmma_bf(afrag, bfrag[fn], acc[fm][fn]);
    }

    wait_async0();
    __syncthreads();
    buf ^= 1;
  }

#pragma unroll
  for (int fm = 0; fm < 4; ++fm)
#pragma unroll
    for (int fn = 0; fn < 2; ++fn)
#pragma unroll
      for (int r = 0; r < 8; ++r) {
        int m = bm + wm + fm * 16 + r + 8 * hlf;
        int n = bn + wn + fn * 16 + l16;
        C[(size_t)m * N + n] = acc[fm][fn][r];
      }
}

// ---------------- rotary + scale + fp32->bf16 ----------------
__global__ __launch_bounds__(128) void rotary_cvt_kernel(const float* __restrict__ X,
                                                         bf16* __restrict__ Y,
                                                         float scale) {
  const int row = blockIdx.x;       // (b*T + t)*H + h
  const int bt = row >> 3;
  const int h = row & 7;
  const int t = bt & (T - 1);
  const int d = threadIdx.x;        // 0..127
  const float* x = X + (size_t)bt * (H * DK) + h * DK;
  bf16* y = Y + (size_t)bt * (H * DK) + h * DK;
  float inv = exp2f(-((float)(2 * d) * (1.0f / 256.0f)) * 13.28771237954945f);
  float fr = (float)t * inv;
  float s, c;
  __sincosf(fr, &s, &c);
  float x1 = x[d], x2 = x[d + 128];
  y[d]       = f2bf((x1 * c - x2 * s) * scale);
  y[d + 128] = f2bf((x2 * c + x1 * s) * scale);
}

// ---------------- chunked retention ----------------
// grid: B*H*4 blocks; each block owns (b, h, 128-wide v slice), 256 threads (8 waves).
__global__ __launch_bounds__(256) void retention_kernel(const bf16* __restrict__ Qbf,
                                                        const bf16* __restrict__ Kbf,
                                                        const float* __restrict__ Vf,
                                                        float* __restrict__ Of) {
  __shared__ float Sdv[128][256];   // state, transposed: [dv][dk], fp32 (128 KB)
  __shared__ bf16 qs[64][264];      // q chunk [t][dk]
  __shared__ bf16 ks[64][264];      // k chunk [t][dk]
  __shared__ bf16 kT[256][72];      // k chunk transposed [dk][t]
  __shared__ bf16 vT[128][72];      // v slice transposed [dv][t]
  __shared__ bf16 Asm[64][72];      // decayed attention A [i][j]

  const int tid = threadIdx.x;
  const int wave = tid >> 5, lane = tid & 31;
  const int hlf = lane >> 4, l16 = lane & 15;

  const int vblk = blockIdx.x & 3;
  const int bh = blockIdx.x >> 2;
  const int b = bh >> 3, h = bh & 7;

  const float gamma = 1.0f - exp2f(-5.0f - (float)h);
  const float l2g = log2f(gamma);
  const float gC = exp2f(64.0f * l2g);

  for (int i = tid; i < 128 * 256; i += 256) (&Sdv[0][0])[i] = 0.0f;
  __syncthreads();

  const size_t rowQ = (size_t)H * DK;
  const size_t rowV = (size_t)H * DV;
  const bf16* qbase = Qbf + (size_t)b * T * rowQ + h * DK;
  const bf16* kbase = Kbf + (size_t)b * T * rowQ + h * DK;
  const float* vbase = Vf + (size_t)b * T * rowV + h * DV + vblk * 128;
  float* obase = Of + (size_t)b * T * rowV + h * DV + vblk * 128;

  for (int c = 0; c < NCHUNK; ++c) {
    const int t0 = c * 64;
    // ---- stage chunk into LDS ----
    {
      const int i = tid >> 2;        // 0..63 (time row)
      const int part = tid & 3;      // quarter of dk / dv
      // q: async DMA straight into LDS (row-major copy)
      const bf16* gq = qbase + (size_t)(t0 + i) * rowQ + part * 64;
#pragma unroll
      for (int j = 0; j < 8; ++j)
        async_b128(lds_addr(&qs[i][part * 64 + j * 8]), gq + j * 8);
      // k: batch-load to regs (needed for kT scatter), then store both forms
      const bf16* gk = kbase + (size_t)(t0 + i) * rowQ + part * 64;
      bf16x8 kv[8];
#pragma unroll
      for (int j = 0; j < 8; ++j) kv[j] = *(const bf16x8*)(gk + j * 8);
#pragma unroll
      for (int j = 0; j < 8; ++j) *(bf16x8*)&ks[i][part * 64 + j * 8] = kv[j];
#pragma unroll
      for (int j = 0; j < 8; ++j)
#pragma unroll
        for (int e = 0; e < 8; ++e) kT[part * 64 + j * 8 + e][i] = kv[j][e];
      // v: batch fp32 loads -> bf16 -> transposed scatter
      const float* gv = vbase + (size_t)(t0 + i) * rowV + part * 32;
      float vv[32];
#pragma unroll
      for (int j = 0; j < 32; ++j) vv[j] = gv[j];
#pragma unroll
      for (int j = 0; j < 32; ++j) vT[part * 32 + j][i] = f2bf(vv[j]);
    }
    wait_async0();
    __syncthreads();

    // ---- A = q @ k^T, decay mask, store bf16 ----
#pragma unroll
    for (int sub = 0; sub < 2; ++sub) {
      const int idx = wave * 2 + sub;       // 16 tiles of 16x16
      const int tm = idx >> 2, tn = idx & 3;
      v8f a = (v8f){0,0,0,0,0,0,0,0};
#pragma unroll
      for (int kk = 0; kk < 8; ++kk) {
        v16bf af = frag_ld(&qs[tm * 16 + l16][0], kk * 32, hlf);
        v16bf bf_ = frag_ld(&ks[tn * 16 + l16][0], kk * 32, hlf);  // B = k^T
        a = wmma_bf(af, bf_, a);
      }
#pragma unroll
      for (int r = 0; r < 8; ++r) {
        int i = tm * 16 + r + 8 * hlf;
        int j = tn * 16 + l16;
        float w = (i >= j) ? exp2f((float)(i - j) * l2g) : 0.0f;
        Asm[i][j] = f2bf(a[r] * w);
      }
    }
    __syncthreads();

    // ---- o = gamma^(i+1) * (q @ S) + (A o D) @ v ----
#pragma unroll
    for (int sub = 0; sub < 4; ++sub) {
      const int idx = wave * 4 + sub;       // 32 tiles (4 x 8)
      const int tm = idx >> 3, tn = idx & 7;
      const int dv = tn * 16 + l16;
      v8f o = (v8f){0,0,0,0,0,0,0,0};
#pragma unroll
      for (int kk = 0; kk < 8; ++kk) {      // inter-chunk: K = DK = 256
        v16bf af = frag_ld(&qs[tm * 16 + l16][0], kk * 32, hlf);
        const float* sp0 = &Sdv[dv][kk * 32 + 8 * hlf];
        const float* sp1 = &Sdv[dv][kk * 32 + 16 + 8 * hlf];
        v16bf bf_;
#pragma unroll
        for (int e = 0; e < 8; ++e) { bf_[e] = f2bf(sp0[e]); bf_[e + 8] = f2bf(sp1[e]); }
        o = wmma_bf(af, bf_, o);
      }
#pragma unroll
      for (int r = 0; r < 8; ++r) {         // row decay gamma^(i+1)
        int i = tm * 16 + r + 8 * hlf;
        o[r] *= exp2f((float)(i + 1) * l2g);
      }
#pragma unroll
      for (int kk = 0; kk < 2; ++kk) {      // intra-chunk: K = 64
        v16bf af = frag_ld(&Asm[tm * 16 + l16][0], kk * 32, hlf);
        v16bf bf_ = frag_ld(&vT[dv][0], kk * 32, hlf);
        o = wmma_bf(af, bf_, o);
      }
#pragma unroll
      for (int r = 0; r < 8; ++r) {
        int i = tm * 16 + r + 8 * hlf;
        obase[(size_t)(t0 + i) * rowV + dv] = o[r];
      }
    }
    __syncthreads();

    // ---- S = gC * S + k^T @ v ----
#pragma unroll
    for (int sub = 0; sub < 16; ++sub) {
      const int idx = wave * 16 + sub;      // 128 tiles (16 x 8)
      const int tm = idx >> 3, tn = idx & 7;
      const int dv = tn * 16 + l16;
      v8f s;
#pragma unroll
      for (int r = 0; r < 8; ++r) {
        int dk = tm * 16 + r + 8 * hlf;
        s[r] = gC * Sdv[dv][dk];
      }
#pragma unroll
      for (int kk = 0; kk < 2; ++kk) {      // K = 64 (time)
        v16bf af = frag_ld(&kT[tm * 16 + l16][0], kk * 32, hlf);
        v16bf bf_ = frag_ld(&vT[dv][0], kk * 32, hlf);
        s = wmma_bf(af, bf_, s);
      }
#pragma unroll
      for (int r = 0; r < 8; ++r) {
        int dk = tm * 16 + r + 8 * hlf;
        Sdv[dv][dk] = s[r];
      }
    }
    __syncthreads();
  }
}

// ---------------- RMSNorm (per head-dim) + swish gate, fp32 -> bf16 ----------------
__global__ __launch_bounds__(128) void gate_norm_kernel(const float* __restrict__ Of,
                                                        const float* __restrict__ Gf,
                                                        const float* __restrict__ gw,
                                                        bf16* __restrict__ Obf) {
  __shared__ float red[4];
  const int row = blockIdx.x;        // (b*T + t)*H + h
  const int bt = row >> 3, h = row & 7;
  const float* o = Of + (size_t)bt * (H * DV) + h * DV;
  const float* g = Gf + (size_t)bt * (H * DV) + h * DV;
  bf16* out = Obf + (size_t)bt * (H * DV) + h * DV;
  const int tid = threadIdx.x;

  float v[4];
  float ss = 0.0f;
#pragma unroll
  for (int j = 0; j < 4; ++j) {
    v[j] = o[tid * 4 + j];
    ss += v[j] * v[j];
  }
#pragma unroll
  for (int off = 16; off > 0; off >>= 1) ss += __shfl_xor(ss, off, 32);
  if ((tid & 31) == 0) red[tid >> 5] = ss;
  __syncthreads();
  float tot = red[0] + red[1] + red[2] + red[3];
  float rinv = rsqrtf(tot * (1.0f / (float)DV) + 1e-5f);
#pragma unroll
  for (int j = 0; j < 4; ++j) {
    int d = tid * 4 + j;
    float gg = g[d];
    float sw = gg / (1.0f + __expf(-gg));
    out[d] = f2bf(v[j] * rinv * gw[d] * sw);
  }
}

// ---------------- host ----------------
static inline size_t alignup(size_t x) { return (x + 255) & ~(size_t)255; }

extern "C" void kernel_launch(void* const* d_in, const int* in_sizes, int n_in,
                              void* d_out, int out_size, void* d_ws, size_t ws_size,
                              hipStream_t stream) {
  const float* hs = (const float*)d_in[0];
  const float* Wq = (const float*)d_in[1];
  const float* Wk = (const float*)d_in[2];
  const float* Wv = (const float*)d_in[3];
  const float* Wg = (const float*)d_in[4];
  const float* Wo = (const float*)d_in[5];
  const float* gw = (const float*)d_in[6];
  float* out = (float*)d_out;

  const size_t BT = (size_t)B * T;        // 4096
  char* p = (char*)d_ws;
  auto carve = [&](size_t bytes) { char* r = p; p += alignup(bytes); return r; };

  bf16* Xbf = (bf16*)carve(BT * HID * 2);
  bf16* WqT = (bf16*)carve((size_t)HID * H * DK * 2);
  bf16* WkT = (bf16*)carve((size_t)HID * H * DK * 2);
  bf16* WvT = (bf16*)carve((size_t)HID * H * DV * 2);
  bf16* WgT = (bf16*)carve((size_t)HID * H * DV * 2);
  bf16* WoT = (bf16*)carve((size_t)H * DV * HID * 2);
  float* Qf = (float*)carve(BT * H * DK * 4);
  float* Kf = (float*)carve(BT * H * DK * 4);
  float* Vf = (float*)carve(BT * H * DV * 4);
  float* Gf = (float*)carve(BT * H * DV * 4);
  bf16* Qbf = (bf16*)carve(BT * H * DK * 2);
  bf16* Kbf = (bf16*)carve(BT * H * DK * 2);
  float* Of = (float*)carve(BT * H * DV * 4);
  bf16* Obf = (bf16*)carve(BT * H * DV * 2);

  // 1) precision prep
  {
    int n = (int)(BT * HID);
    cvt_f2bf_kernel<<<(n + 255) / 256, 256, 0, stream>>>(hs, Xbf, n);
  }
  {
    int n = HID * H * DK;   // 2048*2048
    transpose_cvt_kernel<<<(n + 255) / 256, 256, 0, stream>>>(Wq, WqT, HID, H * DK);
    transpose_cvt_kernel<<<(n + 255) / 256, 256, 0, stream>>>(Wk, WkT, HID, H * DK);
  }
  {
    int n = HID * H * DV;   // 2048*4096
    transpose_cvt_kernel<<<(n + 255) / 256, 256, 0, stream>>>(Wv, WvT, HID, H * DV);
    transpose_cvt_kernel<<<(n + 255) / 256, 256, 0, stream>>>(Wg, WgT, HID, H * DV);
    transpose_cvt_kernel<<<(n + 255) / 256, 256, 0, stream>>>(Wo, WoT, H * DV, HID);
  }

  // 2) projections (bf16 WMMA, fp32 out)
  gemm_bf16_kernel<<<dim3((H * DK) / 128, BT / 128), 256, 0, stream>>>(
      Xbf, WqT, Qf, (int)BT, H * DK, HID);
  gemm_bf16_kernel<<<dim3((H * DK) / 128, BT / 128), 256, 0, stream>>>(
      Xbf, WkT, Kf, (int)BT, H * DK, HID);
  gemm_bf16_kernel<<<dim3((H * DV) / 128, BT / 128), 256, 0, stream>>>(
      Xbf, WvT, Vf, (int)BT, H * DV, HID);
  gemm_bf16_kernel<<<dim3((H * DV) / 128, BT / 128), 256, 0, stream>>>(
      Xbf, WgT, Gf, (int)BT, H * DV, HID);

  // 3) rotary (+ DK^-0.5 on q), back to bf16
  rotary_cvt_kernel<<<(int)(BT * H), 128, 0, stream>>>(Qf, Qbf, 0.0625f);
  rotary_cvt_kernel<<<(int)(BT * H), 128, 0, stream>>>(Kf, Kbf, 1.0f);

  // 4) chunked retention scan
  retention_kernel<<<B * H * 4, 256, 0, stream>>>(Qbf, Kbf, Vf, Of);

  // 5) RMSNorm + swish gate
  gate_norm_kernel<<<(int)(BT * H), 128, 0, stream>>>(Of, Gf, gw, Obf);

  // 6) output projection -> d_out (fp32)
  gemm_bf16_kernel<<<dim3(HID / 128, BT / 128), 256, 0, stream>>>(
      Obf, WoT, out, (int)BT, HID, H * DV);
}
